// GeometricAttention_10505490006638
// MI455X (gfx1250) — matrix operand order repaired and checked by reading, not compiled, not run on hardware
//
#include <hip/hip_runtime.h>

typedef _Float16 h16;
typedef __attribute__((ext_vector_type(2)))  _Float16 v2h;
typedef __attribute__((ext_vector_type(16))) _Float16 v16h;
typedef __attribute__((ext_vector_type(8)))  float    v8f;

#define B_ 4
#define L_ 2048
#define S_ 2048
#define H_ 8
#define E_ 64
#define D_ 64

#define LDH 72   // padded LDS row stride (halves) -> conflict-free gathers

#if __has_builtin(__builtin_amdgcn_sqrtf)
#define FAST_SQRT(x) __builtin_amdgcn_sqrtf(x)
#else
#define FAST_SQRT(x) __builtin_sqrtf(x)
#endif

#if __has_builtin(__builtin_amdgcn_rcpf)
#define FAST_RCP(x) __builtin_amdgcn_rcpf(x)
#else
#define FAST_RCP(x) (1.0f / (x))
#endif

static __device__ __forceinline__ float fast_exp2(float x) {
#if __has_builtin(__builtin_amdgcn_exp2f)
  return __builtin_amdgcn_exp2f(x);
#else
  return exp2f(x);
#endif
}

__global__ __launch_bounds__(128) void geo_attn_kernel(const float* __restrict__ Q,
                                                       const float* __restrict__ K,
                                                       const float* __restrict__ V,
                                                       float* __restrict__ Out) {
  constexpr float EPS = 1e-8f;
  // softmax(u * SCALE): fold SCALE (=1/sqrt(64)) and log2(e) into the exp2 argument
  constexpr float C2 = 0.125f * 1.44269504088896340736f;

  __shared__ h16  Qs[64 * LDH];        // Qs[row][e]
  __shared__ h16  Ks[64 * LDH];        // Ks[key][e]
  __shared__ h16  Vt[64 * LDH];        // Vt[d][key]   (transposed for B-frag gathers)
  __shared__ h16  Ps[4][16 * LDH];     // per-wave P staging (C-layout -> A-layout)
  __shared__ float qn2s[64];
  __shared__ float kn2s[64];

  const int tid  = threadIdx.x;
  const int lane = tid & 31;
  const int wave = tid >> 5;
  const int hi   = (lane >> 4) & 1;    // which 16-lane half
  const int ln   = lane & 15;

  const int bid = blockIdx.x;
  const int mt  = bid & 31;            // L/64 = 32 tiles
  const int h   = (bid >> 5) & 7;      // H = 8
  const int b   = bid >> 8;            // B = 4
  const int q0  = mt * 64;

  // ---- stage Q tile (f32 -> f16, packed cvt) + per-row |q|^2 from rounded values ----
  {
    const int row = tid >> 1;
    const int off = (tid & 1) * 32;
    const float* qr = Q + (size_t)((b * L_ + q0 + row) * H_ + h) * E_ + off;
    float acc = 0.f;
#pragma unroll
    for (int e = 0; e < 16; ++e) {
      h16 x0 = (h16)qr[2 * e];
      h16 x1 = (h16)qr[2 * e + 1];
      *(v2h*)&Qs[row * LDH + off + 2 * e] = (v2h){x0, x1};
      float f0 = (float)x0, f1 = (float)x1;
      acc = fmaf(f0, f0, acc);
      acc = fmaf(f1, f1, acc);
    }
    acc += __shfl_xor(acc, 1, 32);
    if ((tid & 1) == 0) qn2s[row] = acc;
  }
  __syncthreads();

  // ---- Q A-fragments (16x32, CDNA5 A layout), kept in registers for whole S loop ----
  v16h qa[2];
#pragma unroll
  for (int kkI = 0; kkI < 2; ++kkI) {
    const int kk = kkI * 32;
#pragma unroll
    for (int v = 0; v < 8; ++v) {
#pragma unroll
      for (int j = 0; j < 2; ++j) {
        const int k = kk + ((v & 3) * 2 + j) + ((v & 4) ? 16 : 0) + (hi ? 8 : 0);
        qa[kkI][2 * v + j] = Qs[(wave * 16 + ln) * LDH + k];
      }
    }
  }
  float qn2r[8];
#pragma unroll
  for (int r = 0; r < 8; ++r) qn2r[r] = qn2s[wave * 16 + r + hi * 8];

  // all-ones B fragment: P x ones = per-row softmax denominators
  v16h ones;
#pragma unroll
  for (int i = 0; i < 16; ++i) ones[i] = (h16)1.0f;

  // ---- online-softmax running state (max tracked in UNSCALED score units) ----
  float m_run[8], l_run[8];
  v8f acc_o[4];
#pragma unroll
  for (int r = 0; r < 8; ++r) { m_run[r] = -3.0e38f; l_run[r] = 0.f; }
#pragma unroll
  for (int t = 0; t < 4; ++t) acc_o[t] = (v8f){0.f,0.f,0.f,0.f,0.f,0.f,0.f,0.f};

  for (int s0 = 0; s0 < S_; s0 += 64) {
    // ---- stage K tile + |k|^2, V tile transposed ----
    {
      const int row = tid >> 1;
      const int off = (tid & 1) * 32;
      const float* kr = K + (size_t)((b * S_ + s0 + row) * H_ + h) * E_ + off;
      float acc = 0.f;
#pragma unroll
      for (int e = 0; e < 16; ++e) {
        h16 x0 = (h16)kr[2 * e];
        h16 x1 = (h16)kr[2 * e + 1];
        *(v2h*)&Ks[row * LDH + off + 2 * e] = (v2h){x0, x1};
        float f0 = (float)x0, f1 = (float)x1;
        acc = fmaf(f0, f0, acc);
        acc = fmaf(f1, f1, acc);
      }
      acc += __shfl_xor(acc, 1, 32);
      if ((tid & 1) == 0) kn2s[row] = acc;

      const float* vr = V + (size_t)((b * S_ + s0 + row) * H_ + h) * D_ + off;
#pragma unroll
      for (int d = 0; d < 32; ++d) Vt[(off + d) * LDH + row] = (h16)vr[d];

      if (s0 + 64 < S_) {  // gfx1250 global_prefetch of next tile
        __builtin_prefetch(kr + 64 * H_ * E_, 0, 0);
        __builtin_prefetch(vr + 64 * H_ * D_, 0, 0);
      }
    }
    __syncthreads();

    // ---- S_tile = Q x K^T : 4 N-subtiles x 2 k-steps of v_wmma_f32_16x16x32_f16 ----
    v8f c[4];
#pragma unroll
    for (int t = 0; t < 4; ++t) {
      v8f accc = (v8f){0.f,0.f,0.f,0.f,0.f,0.f,0.f,0.f};
#pragma unroll
      for (int kkI = 0; kkI < 2; ++kkI) {
        const int kk = kkI * 32;
        v16h kb;
#pragma unroll
        for (int v = 0; v < 8; ++v) {
#pragma unroll
          for (int j = 0; j < 2; ++j) {
            const int k = kk + ((v & 3) * 2 + j) + ((v & 4) ? 16 : 0) + (hi ? 8 : 0);
            kb[2 * v + j] = Ks[(t * 16 + ln) * LDH + k];   // B[k][n] = K[key=n][e=k]
          }
        }
        accc = __builtin_amdgcn_wmma_f32_16x16x32_f16(false, qa[kkI], false, kb,
                                                      (short)0, accc, false, false);
      }
      c[t] = accc;
    }

    float kn2n[4];
#pragma unroll
    for (int t = 0; t < 4; ++t) kn2n[t] = kn2s[t * 16 + ln];

    // ---- wedge score (ALPHA==1 -> pure wedge) + online softmax ----
    float corr8[8];
#pragma unroll
    for (int r = 0; r < 8; ++r) {
      float mx = -3.0e38f;
#pragma unroll
      for (int t = 0; t < 4; ++t) {
        float dot = c[t][r];
        float w2  = fmaxf(fmaf(-dot, dot, qn2r[r] * kn2n[t]), 0.f);
        float u   = FAST_SQRT(w2 + EPS);   // unscaled score
        c[t][r] = u;
        mx = fmaxf(mx, u);
      }
      mx = fmaxf(mx, __shfl_xor(mx, 1, 32));
      mx = fmaxf(mx, __shfl_xor(mx, 2, 32));
      mx = fmaxf(mx, __shfl_xor(mx, 4, 32));
      mx = fmaxf(mx, __shfl_xor(mx, 8, 32));
      const float mn = fmaxf(m_run[r], mx);
      corr8[r] = fast_exp2((m_run[r] - mn) * C2);
      m_run[r] = mn;
      // exp + f16 convert + stage to per-wave LDS (C-layout -> A-layout) in one pass
#pragma unroll
      for (int t = 0; t < 4; ++t) {
        float p = fast_exp2((c[t][r] - mn) * C2);
        Ps[wave][(r + hi * 8) * LDH + t * 16 + ln] = (h16)p;
      }
#pragma unroll
      for (int t = 0; t < 4; ++t) acc_o[t][r] *= corr8[r];
    }
    asm volatile("s_wait_dscnt 0" ::: "memory");  // wave-internal LDS RAW ordering

    // ---- P A-fragments, row sums via WMMA (P x ones), then O += P x V ----
    v16h pa[2];
#pragma unroll
    for (int kkI = 0; kkI < 2; ++kkI) {
      const int kk = kkI * 32;
#pragma unroll
      for (int v = 0; v < 8; ++v) {
#pragma unroll
        for (int j = 0; j < 2; ++j) {
          const int k = kk + ((v & 3) * 2 + j) + ((v & 4) ? 16 : 0) + (hi ? 8 : 0);
          pa[kkI][2 * v + j] = Ps[wave][ln * LDH + k];
        }
      }
    }
    v8f csum = (v8f){0.f,0.f,0.f,0.f,0.f,0.f,0.f,0.f};
    csum = __builtin_amdgcn_wmma_f32_16x16x32_f16(false, pa[0], false, ones,
                                                  (short)0, csum, false, false);
    csum = __builtin_amdgcn_wmma_f32_16x16x32_f16(false, pa[1], false, ones,
                                                  (short)0, csum, false, false);

#pragma unroll
    for (int kkI = 0; kkI < 2; ++kkI) {
      const int kk = kkI * 32;
#pragma unroll
      for (int t = 0; t < 4; ++t) {
        v16h vb;
#pragma unroll
        for (int v = 0; v < 8; ++v) {
#pragma unroll
          for (int j = 0; j < 2; ++j) {
            const int k = kk + ((v & 3) * 2 + j) + ((v & 4) ? 16 : 0) + (hi ? 8 : 0);
            vb[2 * v + j] = Vt[(t * 16 + ln) * LDH + k];  // B[k][n] = V[key=k][d=n]
          }
        }
        acc_o[t] = __builtin_amdgcn_wmma_f32_16x16x32_f16(false, pa[kkI], false, vb,
                                                          (short)0, acc_o[t], false, false);
      }
    }

#pragma unroll
    for (int r = 0; r < 8; ++r) l_run[r] = fmaf(l_run[r], corr8[r], csum[r]);

    __syncthreads();  // protect Ks/Vt before next tile's overwrite
  }

  // ---- epilogue: normalize (fast rcp) and store [B,L,H,D] ----
  float inv_l[8];
#pragma unroll
  for (int r = 0; r < 8; ++r) inv_l[r] = FAST_RCP(l_run[r]);
#pragma unroll
  for (int t = 0; t < 4; ++t) {
#pragma unroll
    for (int r = 0; r < 8; ++r) {
      const int row  = q0 + wave * 16 + r + hi * 8;
      const int dcol = t * 16 + ln;
      Out[(size_t)((b * L_ + row) * H_ + h) * D_ + dcol] = acc_o[t][r] * inv_l[r];
    }
  }
}

extern "C" void kernel_launch(void* const* d_in, const int* in_sizes, int n_in,
                              void* d_out, int out_size, void* d_ws, size_t ws_size,
                              hipStream_t stream) {
  (void)in_sizes; (void)n_in; (void)out_size; (void)d_ws; (void)ws_size;
  const float* Q = (const float*)d_in[0];
  const float* K = (const float*)d_in[1];
  const float* V = (const float*)d_in[2];
  float* Out = (float*)d_out;
  // grid: (L/64) * H * B = 32 * 8 * 4 = 1024 workgroups, 4 wave32 each
  geo_attn_kernel<<<dim3(1024), dim3(128), 0, stream>>>(Q, K, V, Out);
}
